// CrossMambaFusion_65360812311159
// MI455X (gfx1250) — matrix operand order repaired and checked by reading, not compiled, not run on hardware
//
#include <hip/hip_runtime.h>
#include <hip/hip_bf16.h>

typedef __bf16 bf16;
typedef __attribute__((ext_vector_type(16))) __bf16 v16bf;
typedef __attribute__((ext_vector_type(8)))  float  v8f;
typedef uint32_t v4u __attribute__((ext_vector_type(4)));
typedef uint32_t v8u __attribute__((ext_vector_type(8)));

#define C_DIM 256
#define DI    512
#define NST   16
#define RRK   32
#define LSEQ  4096
#define NB    2
#define BL    8192   // NB*LSEQ

__device__ __forceinline__ float sigm(float x) { return 1.0f / (1.0f + __expf(-x)); }
__device__ __forceinline__ float silu(float x) { return x * sigm(x); }

// ---- WMMA fragment helpers (ISA 7.12.2 layouts, wave32) -------------------
// A 16x32 bf16: lane m=lane&15; K chunks [kb..kb+7] and [16+kb..16+kb+7], kb=(lane>>4)*8
__device__ __forceinline__ v16bf load_a16(const bf16* A, int lda, int m0, int k0, int lane) {
  const int m  = lane & 15;
  const int kb = (lane >> 4) << 3;
  const bf16* p0 = A + (size_t)(m0 + m) * lda + k0 + kb;
  v16bf a;
#pragma unroll
  for (int j = 0; j < 8; ++j) a[j] = p0[j];
#pragma unroll
  for (int j = 0; j < 8; ++j) a[8 + j] = p0[16 + j];
  return a;
}
// B 32x16 bf16, pre-packed: 16 contiguous bf16 per lane per (nt,kt) tile
__device__ __forceinline__ v16bf load_bp(const bf16* Bp, int nt, int kt, int KT, int lane) {
  const bf16* p = Bp + ((((size_t)nt * KT + kt) * 32 + lane) << 4);
  v16bf b;
#pragma unroll
  for (int j = 0; j < 16; ++j) b[j] = p[j];
  return b;
}
__device__ __forceinline__ v8f wmma_bf(v16bf a, v16bf b, v8f c) {
  return __builtin_amdgcn_wmma_f32_16x16x32_bf16(false, a, false, b, (short)0, c, false, false);
}

// ---- weight repack: row-major f32 [K,N] -> fragment-ordered bf16 ----------
__global__ void pack_b_kernel(const float* __restrict__ src, bf16* __restrict__ dst,
                              int K, int Nn) {
  int idx = blockIdx.x * 256 + threadIdx.x;
  if (idx >= K * Nn) return;
  int j    = idx & 15;
  int lane = (idx >> 4) & 31;
  int t    = idx >> 9;
  int KT   = K >> 5;
  int kt   = t % KT;
  int nt   = t / KT;
  int k = kt * 32 + ((lane >> 4) << 3) + (j & 7) + ((j >> 3) << 4);
  int n = nt * 16 + (lane & 15);
  dst[idx] = (bf16)src[(size_t)k * Nn + n];
}

// ---- NCHW -> (BL, C) transpose, bf16 (+optional f32 copy) -----------------
__global__ void transpose_in_kernel(const float* __restrict__ src,
                                    bf16* __restrict__ dst_bf,
                                    float* __restrict__ dst_f, int write_f32) {
  __shared__ float tile[32][C_DIM + 1];
  int b   = blockIdx.x >> 7;          // 128 hw-tiles per batch
  int hw0 = (blockIdx.x & 127) << 5;
  int tid = threadIdx.x;
  int hw_l = tid & 31, c0 = tid >> 5;
#pragma unroll 4
  for (int cc = 0; cc < 32; ++cc) {
    int c = cc * 8 + c0;
    tile[hw_l][c] = src[((size_t)(b * C_DIM + c) << 12) + hw0 + hw_l];
  }
  __syncthreads();
#pragma unroll 4
  for (int i = 0; i < 32; ++i) {
    size_t bl = (size_t)b * LSEQ + hw0 + i;
    float v = tile[i][tid];
    dst_bf[bl * C_DIM + tid] = (bf16)v;
    if (write_f32) dst_f[bl * C_DIM + tid] = v;
  }
}

// ---- K1: dec/enc projections + cross gating -------------------------------
__global__ void proj_fuse_kernel(const bf16* __restrict__ decA, const bf16* __restrict__ encA,
                                 const bf16* __restrict__ decWp, const bf16* __restrict__ encWp,
                                 const float* __restrict__ dec_b, const float* __restrict__ enc_b,
                                 bf16* __restrict__ comb_bf, float* __restrict__ sgate) {
  int lane = threadIdx.x & 31;
  int tile = blockIdx.x * 8 + (threadIdx.x >> 5);
  int rt = tile >> 5, nt = tile & 31;          // 512 row tiles x 32 col tiles
  v8f a1 = {}, a2 = {}, a3 = {};
#pragma unroll
  for (int kt = 0; kt < 8; ++kt) {             // K = 256
    v16bf ad = load_a16(decA, C_DIM, rt * 16, kt * 32, lane);
    v16bf ae = load_a16(encA, C_DIM, rt * 16, kt * 32, lane);
    a1 = wmma_bf(ad, load_bp(decWp, nt,      kt, 8, lane), a1);  // dec_x
    a2 = wmma_bf(ae, load_bp(encWp, nt,      kt, 8, lane), a2);  // enc_p
    a3 = wmma_bf(ad, load_bp(decWp, nt + 32, kt, 8, lane), a3);  // dec_gate
  }
  int n = nt * 16 + (lane & 15);
  int mh = (lane >> 4) << 3;
#pragma unroll
  for (int v = 0; v < 8; ++v) {
    size_t row = (size_t)rt * 16 + mh + v;
    float c1 = a1[v] + dec_b[n];
    float c2 = a2[v] + enc_b[n];
    float c3 = a3[v] + dec_b[DI + n];
    comb_bf[row * DI + n] = (bf16)(c1 * sigm(c2) + c2);
    sgate[row * DI + n]   = sigm(c3);
  }
}

// ---- K2: in_w GEMM -> xm (f32) and silu(z); A panel staged via TDM --------
__global__ void in_gemm_kernel(const bf16* __restrict__ A, const bf16* __restrict__ Wp,
                               const float* __restrict__ in_b,
                               float* __restrict__ xm, float* __restrict__ sz) {
  __shared__ bf16 sA[16 * DI];                 // 16 KB A row-panel
  int lane = threadIdx.x & 31;
  int wave = threadIdx.x >> 5;
  int tile = blockIdx.x * 8 + wave;
  int rt = tile >> 6, nt = tile & 63;          // 8 waves of a block share rt

  if (wave == 0) {
    // 2D Tensor DMA descriptor (ISA cdna5 ch.8): 16x512 bf16 tile -> LDS
    uint64_t gaddr = (uint64_t)(uintptr_t)(A + (size_t)rt * 16 * DI);
    uint32_t laddr = (uint32_t)(uintptr_t)(&sA[0]);
    v4u g0;
    g0[0] = 1u;                                              // count=1, user D#
    g0[1] = laddr;                                           // lds_addr
    g0[2] = (uint32_t)gaddr;                                 // global_addr[31:0]
    g0[3] = (uint32_t)((gaddr >> 32) & 0x1FFFFFFu) | (2u << 30); // addr[56:32] | type=2
    v8u g1;
    g1[0] = 1u << 16;                                        // data_size=1 (2 bytes)
    g1[1] = ((uint32_t)DI & 0xFFFFu) << 16;                  // tensor_dim0[15:0]
    g1[2] = ((uint32_t)DI >> 16) | (((uint32_t)BL & 0xFFFFu) << 16); // td0 hi | td1 lo
    g1[3] = ((uint32_t)BL >> 16) | ((uint32_t)DI << 16);     // td1 hi | tile_dim0=512
    g1[4] = 16u;                                             // tile_dim1=16, tile_dim2=0
    g1[5] = (uint32_t)DI;                                    // tensor_dim0_stride[31:0]
    g1[6] = 0u;                                              // stride hi | dim1_stride lo
    g1[7] = 0u;
    asm volatile("tensor_load_to_lds %0, %1" :: "s"(g0), "s"(g1) : "memory");
    __builtin_amdgcn_s_wait_tensorcnt(0);
  }
  __syncthreads();

  v8f acc = {};
#pragma unroll
  for (int kt = 0; kt < 16; ++kt) {            // K = 512
    v16bf a = load_a16(sA, DI, 0, kt * 32, lane);   // ds_load from staged panel
    v16bf b = load_bp(Wp, nt, kt, 16, lane);
    if (kt < 15) __builtin_prefetch(Wp + (((size_t)nt * 16 + kt + 1) * 32 + lane) * 16, 0, 1);
    acc = wmma_bf(a, b, acc);
  }
  int n = nt * 16 + (lane & 15);
  int mh = (lane >> 4) << 3;
#pragma unroll
  for (int v = 0; v < 8; ++v) {
    size_t row = (size_t)rt * 16 + mh + v;
    float c = acc[v] + in_b[n];
    if (n < DI) xm[row * DI + n] = c;
    else        sz[row * DI + (n - DI)] = silu(c);
  }
}

// ---- K3: depthwise causal conv (K=4) + silu -------------------------------
__global__ void conv_silu_kernel(const float* __restrict__ xm, const float* __restrict__ cw,
                                 const float* __restrict__ cb,
                                 float* __restrict__ u, bf16* __restrict__ u_bf) {
  size_t idx = (size_t)blockIdx.x * 256 + threadIdx.x;   // BL*DI threads
  int d  = idx & (DI - 1);
  size_t bl = idx >> 9;
  int l  = bl & (LSEQ - 1);
  int b  = bl >> 12;
  float acc = cb[d];
#pragma unroll
  for (int j = 0; j < 4; ++j) {
    int ls = l - 3 + j;
    if (ls >= 0) acc += xm[((size_t)(b * LSEQ + ls)) * DI + d] * cw[d * 4 + j];
  }
  float uv = silu(acc);
  u[idx] = uv;
  u_bf[idx] = (bf16)uv;
}

// ---- K4: x_proj GEMM (no bias) -> x_dbl f32, first 32 cols also bf16 ------
__global__ void xproj_gemm_kernel(const bf16* __restrict__ A, const bf16* __restrict__ Wp,
                                  float* __restrict__ xdbl, bf16* __restrict__ dtx) {
  int lane = threadIdx.x & 31;
  int tile = blockIdx.x * 8 + (threadIdx.x >> 5);
  int rt = tile >> 2, nt = tile & 3;           // 512 x 4 tiles
  v8f acc = {};
#pragma unroll
  for (int kt = 0; kt < 16; ++kt)
    acc = wmma_bf(load_a16(A, DI, rt * 16, kt * 32, lane), load_bp(Wp, nt, kt, 16, lane), acc);
  int n = nt * 16 + (lane & 15);
  int mh = (lane >> 4) << 3;
#pragma unroll
  for (int v = 0; v < 8; ++v) {
    size_t row = (size_t)rt * 16 + mh + v;
    float c = acc[v];
    xdbl[row * 64 + n] = c;
    if (n < RRK) dtx[row * RRK + n] = (bf16)c;
  }
}

// ---- K5: dt GEMM (K=32 -> one WMMA) + softplus ----------------------------
__global__ void dt_gemm_kernel(const bf16* __restrict__ A, const bf16* __restrict__ Wp,
                               const float* __restrict__ dt_b, float* __restrict__ dt) {
  int lane = threadIdx.x & 31;
  int tile = blockIdx.x * 8 + (threadIdx.x >> 5);
  int rt = tile >> 5, nt = tile & 31;          // 512 x 32 tiles
  v8f acc = {};
  acc = wmma_bf(load_a16(A, RRK, rt * 16, 0, lane), load_bp(Wp, nt, 0, 1, lane), acc);
  int n = nt * 16 + (lane & 15);
  int mh = (lane >> 4) << 3;
#pragma unroll
  for (int v = 0; v < 8; ++v) {
    size_t row = (size_t)rt * 16 + mh + v;
    float c = acc[v] + dt_b[n];
    dt[row * DI + n] = (c > 20.0f) ? c : log1pf(__expf(c));  // softplus
  }
}

// ---- K6: selective scan; wave = 2 channels, lane-per-state ----------------
__global__ void scan_kernel(const float* __restrict__ dt, const float* __restrict__ u,
                            const float* __restrict__ xdbl, const float* __restrict__ sz,
                            const float* __restrict__ A_log, const float* __restrict__ Dp,
                            bf16* __restrict__ ymul) {
  int lane = threadIdx.x & 31;
  int widx = blockIdx.x * 8 + (threadIdx.x >> 5);   // 512 waves
  int b = widx >> 8;
  int d = ((widx & 255) << 1) + (lane >> 4);
  int n = lane & 15;
  float An = -__expf(A_log[d * NST + n]);
  float Dd = Dp[d];
  float h = 0.0f;
  for (int l = 0; l < LSEQ; ++l) {
    size_t bl = (size_t)b * LSEQ + l;
    float dtv = dt[bl * DI + d];
    float uv  = u[bl * DI + d];
    float Bv  = xdbl[bl * 64 + 32 + n];
    float Cv  = xdbl[bl * 64 + 48 + n];
    h = h * __expf(dtv * An) + (dtv * uv) * Bv;
    float y = h * Cv;
#pragma unroll
    for (int m = 8; m >= 1; m >>= 1) y += __shfl_xor(y, m, 32);
    if (n == 0) {
      float yt = y + uv * Dd;
      ymul[bl * DI + d] = (bf16)(yt * sz[bl * DI + d]);
    }
  }
}

// ---- K7: m_out GEMM, fused * sigmoid(dec_gate) ----------------------------
__global__ void mout_gemm_kernel(const bf16* __restrict__ A, const bf16* __restrict__ Wp,
                                 const float* __restrict__ mb, const float* __restrict__ sgate,
                                 bf16* __restrict__ pg) {
  int lane = threadIdx.x & 31;
  int tile = blockIdx.x * 8 + (threadIdx.x >> 5);
  int rt = tile >> 5, nt = tile & 31;          // 512 x 32 tiles
  v8f acc = {};
#pragma unroll
  for (int kt = 0; kt < 16; ++kt)
    acc = wmma_bf(load_a16(A, DI, rt * 16, kt * 32, lane), load_bp(Wp, nt, kt, 16, lane), acc);
  int n = nt * 16 + (lane & 15);
  int mh = (lane >> 4) << 3;
#pragma unroll
  for (int v = 0; v < 8; ++v) {
    size_t row = (size_t)rt * 16 + mh + v;
    pg[row * DI + n] = (bf16)((acc[v] + mb[n]) * sgate[row * DI + n]);
  }
}

// ---- K8: out_w GEMM + residual + LayerNorm (block owns 16 full rows) ------
__global__ void out_ln_kernel(const bf16* __restrict__ A, const bf16* __restrict__ Wp,
                              const float* __restrict__ ob, const float* __restrict__ dec_f,
                              const float* __restrict__ ln_g, const float* __restrict__ ln_b,
                              float* __restrict__ res) {
  __shared__ float sC[16][C_DIM + 8];
  int lane = threadIdx.x & 31;
  int wave = threadIdx.x >> 5;
  int r0 = blockIdx.x * 16;
  v8f a0 = {}, a1 = {};
  int nt0 = wave * 2;
#pragma unroll
  for (int kt = 0; kt < 16; ++kt) {
    v16bf a = load_a16(A, DI, r0, kt * 32, lane);
    a0 = wmma_bf(a, load_bp(Wp, nt0,     kt, 16, lane), a0);
    a1 = wmma_bf(a, load_bp(Wp, nt0 + 1, kt, 16, lane), a1);
  }
  int ln = lane & 15;
  int mh = (lane >> 4) << 3;
#pragma unroll
  for (int v = 0; v < 8; ++v) {
    sC[mh + v][nt0 * 16 + ln]       = a0[v] + ob[nt0 * 16 + ln];
    sC[mh + v][(nt0 + 1) * 16 + ln] = a1[v] + ob[(nt0 + 1) * 16 + ln];
  }
  __syncthreads();
#pragma unroll
  for (int rr = 0; rr < 2; ++rr) {
    int r = wave * 2 + rr;
    size_t bl = (size_t)r0 + r;
    float vals[8], s = 0.0f, s2 = 0.0f;
#pragma unroll
    for (int i = 0; i < 8; ++i) {
      int c = i * 32 + lane;
      float x = sC[r][c] + dec_f[bl * C_DIM + c];
      vals[i] = x; s += x; s2 += x * x;
    }
#pragma unroll
    for (int m = 16; m >= 1; m >>= 1) { s += __shfl_xor(s, m, 32); s2 += __shfl_xor(s2, m, 32); }
    float mu = s * (1.0f / C_DIM);
    float var = s2 * (1.0f / C_DIM) - mu * mu;
    float rstd = rsqrtf(var + 1e-5f);
#pragma unroll
    for (int i = 0; i < 8; ++i) {
      int c = i * 32 + lane;
      res[bl * C_DIM + c] = (vals[i] - mu) * rstd * ln_g[c] + ln_b[c];
    }
  }
}

// ---- K9: (BL,C) -> NCHW ---------------------------------------------------
__global__ void transpose_out_kernel(const float* __restrict__ res, float* __restrict__ out) {
  __shared__ float tile[32][C_DIM + 1];
  int b   = blockIdx.x >> 7;
  int hw0 = (blockIdx.x & 127) << 5;
  int tid = threadIdx.x;
  int hw_l = tid & 31, c0 = tid >> 5;
#pragma unroll 4
  for (int i = 0; i < 32; ++i)
    tile[i][tid] = res[((size_t)b * LSEQ + hw0 + i) * C_DIM + tid];
  __syncthreads();
#pragma unroll 4
  for (int cc = 0; cc < 32; ++cc) {
    int c = cc * 8 + c0;
    out[((size_t)(b * C_DIM + c) << 12) + hw0 + hw_l] = tile[hw_l][c];
  }
}

extern "C" void kernel_launch(void* const* d_in, const int* in_sizes, int n_in,
                              void* d_out, int out_size, void* d_ws, size_t ws_size,
                              hipStream_t stream) {
  const float* decoder_feat = (const float*)d_in[0];
  const float* encoder_feat = (const float*)d_in[1];
  const float* dec_w  = (const float*)d_in[2];
  const float* dec_b  = (const float*)d_in[3];
  const float* enc_w  = (const float*)d_in[4];
  const float* enc_b  = (const float*)d_in[5];
  const float* out_w  = (const float*)d_in[6];
  const float* out_b  = (const float*)d_in[7];
  const float* ln_g   = (const float*)d_in[8];
  const float* ln_b   = (const float*)d_in[9];
  const float* in_w   = (const float*)d_in[10];
  const float* in_b   = (const float*)d_in[11];
  const float* conv_w = (const float*)d_in[12];
  const float* conv_b = (const float*)d_in[13];
  const float* xp_w   = (const float*)d_in[14];
  const float* dt_w   = (const float*)d_in[15];
  const float* dt_b   = (const float*)d_in[16];
  const float* A_log  = (const float*)d_in[17];
  const float* D_par  = (const float*)d_in[18];
  const float* mo_w   = (const float*)d_in[19];
  const float* mo_b   = (const float*)d_in[20];
  float* out = (float*)d_out;

  char* ws = (char*)d_ws;
  size_t off = 0;
  auto alloc = [&](size_t bytes) -> void* {
    void* p = ws + off; off += (bytes + 255) & ~(size_t)255; return p;
  };
  bf16*  dec_bf  = (bf16*)alloc((size_t)BL * C_DIM * 2);
  float* dec_f   = (float*)alloc((size_t)BL * C_DIM * 4);
  bf16*  enc_bf  = (bf16*)alloc((size_t)BL * C_DIM * 2);
  bf16*  comb_bf = (bf16*)alloc((size_t)BL * DI * 2);
  float* sgate   = (float*)alloc((size_t)BL * DI * 4);
  float* xm      = (float*)alloc((size_t)BL * DI * 4);
  float* sz      = (float*)alloc((size_t)BL * DI * 4);
  float* u       = (float*)alloc((size_t)BL * DI * 4);
  bf16*  u_bf    = (bf16*)alloc((size_t)BL * DI * 2);
  float* xdbl    = (float*)alloc((size_t)BL * 64 * 4);
  bf16*  dtx     = (bf16*)alloc((size_t)BL * RRK * 2);
  float* dtv     = (float*)alloc((size_t)BL * DI * 4);
  bf16*  ymul    = (bf16*)alloc((size_t)BL * DI * 2);
  bf16*  pg      = (bf16*)alloc((size_t)BL * DI * 2);
  float* res     = (float*)alloc((size_t)BL * C_DIM * 4);
  bf16* dec_wp = (bf16*)alloc((size_t)C_DIM * 1024 * 2);
  bf16* enc_wp = (bf16*)alloc((size_t)C_DIM * DI * 2);
  bf16* in_wp  = (bf16*)alloc((size_t)DI * 1024 * 2);
  bf16* xp_wp  = (bf16*)alloc((size_t)DI * 64 * 2);
  bf16* dt_wp  = (bf16*)alloc((size_t)RRK * DI * 2);
  bf16* mo_wp  = (bf16*)alloc((size_t)DI * DI * 2);
  bf16* ow_wp  = (bf16*)alloc((size_t)DI * C_DIM * 2);

  // weight repacks (fragment order)
  pack_b_kernel<<<(C_DIM * 1024) / 256, 256, 0, stream>>>(dec_w, dec_wp, C_DIM, 1024);
  pack_b_kernel<<<(C_DIM * DI)  / 256, 256, 0, stream>>>(enc_w, enc_wp, C_DIM, DI);
  pack_b_kernel<<<(DI * 1024)   / 256, 256, 0, stream>>>(in_w,  in_wp,  DI, 1024);
  pack_b_kernel<<<(DI * 64)     / 256, 256, 0, stream>>>(xp_w,  xp_wp,  DI, 64);
  pack_b_kernel<<<(RRK * DI)    / 256, 256, 0, stream>>>(dt_w,  dt_wp,  RRK, DI);
  pack_b_kernel<<<(DI * DI)     / 256, 256, 0, stream>>>(mo_w,  mo_wp,  DI, DI);
  pack_b_kernel<<<(DI * C_DIM)  / 256, 256, 0, stream>>>(out_w, ow_wp,  DI, C_DIM);

  transpose_in_kernel<<<NB * 128, 256, 0, stream>>>(decoder_feat, dec_bf, dec_f, 1);
  transpose_in_kernel<<<NB * 128, 256, 0, stream>>>(encoder_feat, enc_bf, dec_f, 0);

  proj_fuse_kernel<<<(512 * 32) / 8, 256, 0, stream>>>(dec_bf, enc_bf, dec_wp, enc_wp,
                                                       dec_b, enc_b, comb_bf, sgate);
  in_gemm_kernel<<<(512 * 64) / 8, 256, 0, stream>>>(comb_bf, in_wp, in_b, xm, sz);
  conv_silu_kernel<<<((size_t)BL * DI) / 256, 256, 0, stream>>>(xm, conv_w, conv_b, u, u_bf);
  xproj_gemm_kernel<<<(512 * 4) / 8, 256, 0, stream>>>(u_bf, xp_wp, xdbl, dtx);
  dt_gemm_kernel<<<(512 * 32) / 8, 256, 0, stream>>>(dtx, dt_wp, dt_b, dtv);
  scan_kernel<<<512 / 8, 256, 0, stream>>>(dtv, u, xdbl, sz, A_log, D_par, ymul);
  mout_gemm_kernel<<<(512 * 32) / 8, 256, 0, stream>>>(ymul, mo_wp, mo_b, sgate, pg);
  out_ln_kernel<<<BL / 16, 256, 0, stream>>>(pg, ow_wp, out_b, dec_f, ln_g, ln_b, res);
  transpose_out_kernel<<<NB * 128, 256, 0, stream>>>(res, out);
}
